// FastAttention_17222818857532
// MI455X (gfx1250) — compile-verified, matrix operand order checked by reading, not run on hardware
//
#include <hip/hip_runtime.h>
#include <hip/hip_bf16.h>
#include <math.h>

// ---------------------------------------------------------------------------
// Performer-style conditioned causal linear attention, fused, for gfx1250.
// b=4 h=8 n=4096 d=64 m=256, COND_LEN=256, rest processed in 60 chunks of 64.
// All GEMMs via v_wmma_f32_16x16x32_f16 (wave32). All fragment loads are
// contiguous 16B LDS reads (ds_load_b128); the key feature map writes both
// row-major and transposed copies so no strided fragment loads remain.
// ---------------------------------------------------------------------------

typedef __attribute__((ext_vector_type(16))) _Float16 v16h;
typedef __attribute__((ext_vector_type(8)))  float    v8f;

#define B_    4
#define H_    8
#define N_    4096
#define D_    64
#define M_    256
#define L_    256
#define C_    64
#define NCH_  60                       // (4096-256)/64
#define CNORM 0.35355339059327373f     // 64^-0.25
#define RATIO 0.0625f                  // 256^-0.5
#define DIAGF 0.0625f                  // (64^-0.25)^2 * 0.5
#define KEPS  1e-4f
#define SEPS  1e-6f

// order-preserving float<->uint map for atomic max reductions
__device__ __forceinline__ unsigned fkey(float f) {
  int b = __float_as_int(f);
  return (unsigned)(b ^ ((b >> 31) | 0x80000000));
}
__device__ __forceinline__ float fdec(unsigned u) {
  int b = (u & 0x80000000u) ? (int)(u ^ 0x80000000u) : ~(int)u;
  return __int_as_float(b);
}

__device__ __forceinline__ v8f wmma_f16(v16h a, v16h b, v8f c) {
  return __builtin_amdgcn_wmma_f32_16x16x32_f16(false, a, false, b, (short)0, c,
                                                false, false);
}

// A-fragment (16x32 f16) from row-major LDS matrix (row stride ld).
// Also used for B-fragments by passing B^T stored row-major.
__device__ __forceinline__ v16h ld_frag(const _Float16* base, int ld, int r0,
                                        int k0, int lane) {
  const _Float16* p = base + (r0 + (lane & 15)) * ld + k0 + ((lane >> 4) << 3);
  v16h f;
#pragma unroll
  for (int i = 0; i < 8; ++i) { f[i] = p[i]; f[8 + i] = p[16 + i]; }
  return f;
}

// Compute phi(chunk of 64 rows of src) -> dst (64 x 256 f16), and optionally
// a transposed copy dstT (256 x 64 f16, for use as a row-major A operand).
// dash = (CNORM*x) @ P^T via WMMA; stab = per-row max (query) or gstab (key).
__device__ void featuremap(const float* __restrict__ src, _Float16* dst,
                           _Float16* dstT, const _Float16* Pl, _Float16* Xf,
                           float* diag, unsigned* rowkey, bool isQuery,
                           float gstab, int tid) {
  const int lane = tid & 31, wave = tid >> 5;
  if (tid < C_) {
    const float* row = src + tid * D_;
    float s = 0.f;
#pragma unroll 8
    for (int c = 0; c < D_; ++c) {
      float x = row[c];
      s += x * x;
      Xf[tid * D_ + c] = (_Float16)(x * CNORM);
    }
    diag[tid]   = s * DIAGF;
    rowkey[tid] = 0u;
  }
  __syncthreads();
  if (isQuery) {  // pass 1: row max of dash
    for (int t = 0; t < 8; ++t) {
      int tile = wave * 8 + t;
      int r0 = (tile & 3) * 16, c0 = (tile >> 2) * 16;
      v8f acc = {};
      acc = wmma_f16(ld_frag(Xf, D_, r0, 0, lane),  ld_frag(Pl, D_, c0, 0, lane),  acc);
      acc = wmma_f16(ld_frag(Xf, D_, r0, 32, lane), ld_frag(Pl, D_, c0, 32, lane), acc);
#pragma unroll
      for (int r = 0; r < 8; ++r)
        atomicMax(&rowkey[r0 + r + ((lane >> 4) << 3)], fkey(acc[r]));
    }
    __syncthreads();
  }
  // pass 2: recompute dash (cheap) and apply exp with known stab
  for (int t = 0; t < 8; ++t) {
    int tile = wave * 8 + t;
    int r0 = (tile & 3) * 16, c0 = (tile >> 2) * 16;
    v8f acc = {};
    acc = wmma_f16(ld_frag(Xf, D_, r0, 0, lane),  ld_frag(Pl, D_, c0, 0, lane),  acc);
    acc = wmma_f16(ld_frag(Xf, D_, r0, 32, lane), ld_frag(Pl, D_, c0, 32, lane), acc);
#pragma unroll
    for (int r = 0; r < 8; ++r) {
      int row = r0 + r + ((lane >> 4) << 3);
      int col = c0 + (lane & 15);
      float stab = isQuery ? fdec(rowkey[row]) : gstab;
      _Float16 h = (_Float16)(RATIO * (expf(acc[r] - diag[row] - stab) + KEPS));
      dst[row * M_ + col] = h;
      if (dstT) dstT[col * C_ + row] = h;
    }
  }
  __syncthreads();
}

// ---------------------------------------------------------------------------
// Kernel 0: init global-max cell
__global__ void fa_init_kernel(unsigned* key) {
  if (threadIdx.x == 0 && blockIdx.x == 0) *key = 0u;  // == most negative
}

// Kernel 1: global max over all k_dash (stab for keys). 2048 WGs, WMMA.
__global__ void fa_kmax_kernel(const float* __restrict__ k,
                               const float* __restrict__ proj, unsigned* key) {
  __shared__ _Float16 Pl[M_ * D_];
  __shared__ _Float16 Xf[C_ * D_];
  __shared__ float    red[256];
  const int tid = threadIdx.x, lane = tid & 31, wave = tid >> 5;
  const float* kc = k + (size_t)blockIdx.x * C_ * D_;
  for (int i = tid; i < M_ * D_; i += 256) Pl[i] = (_Float16)proj[i];
  if (tid < C_) {
    const float* row = kc + tid * D_;
#pragma unroll 8
    for (int c = 0; c < D_; ++c) Xf[tid * D_ + c] = (_Float16)(row[c] * CNORM);
  }
  __syncthreads();
  float m = -3.4e38f;
  for (int t = 0; t < 8; ++t) {
    int tile = wave * 8 + t;
    int r0 = (tile & 3) * 16, c0 = (tile >> 2) * 16;
    v8f acc = {};
    acc = wmma_f16(ld_frag(Xf, D_, r0, 0, lane),  ld_frag(Pl, D_, c0, 0, lane),  acc);
    acc = wmma_f16(ld_frag(Xf, D_, r0, 32, lane), ld_frag(Pl, D_, c0, 32, lane), acc);
#pragma unroll
    for (int r = 0; r < 8; ++r) m = fmaxf(m, acc[r]);
  }
  red[tid] = m;
  __syncthreads();
  for (int s = 128; s > 0; s >>= 1) {
    if (tid < s) red[tid] = fmaxf(red[tid], red[tid + s]);
    __syncthreads();
  }
  if (tid == 0) atomicMax(key, fkey(red[0]));
}

// Kernel 2: fused attention; one workgroup (8 wave32) per (b,h).
__global__ void fa_attn_kernel(const float* __restrict__ q,
                               const float* __restrict__ k,
                               const float* __restrict__ v,
                               const float* __restrict__ proj,
                               float* __restrict__ out,
                               const unsigned* __restrict__ kmax_key) {
  extern __shared__ char smem[];
  _Float16* Pl = (_Float16*)smem;       // 256x64   projection (f16)
  _Float16* Qp = Pl + M_ * D_;          // 64x256   phi(q) chunk
  _Float16* Kp = Qp + C_ * M_;          // 64x256   phi(k) chunk
  _Float16* KT = Kp + C_ * M_;          // 256x64   phi(k)^T chunk
  _Float16* ST = KT + M_ * C_;          // 64x256   S^T (f16 copy for WMMA B)
  _Float16* VT = ST + D_ * M_;          // 64x64    V^T chunk
  _Float16* Af = VT + D_ * C_;          // 64x64    X scratch / masked A
  float*    S    = (float*)(Af + C_ * C_);  // 256x64 state (f32 accumulator)
  float*    z    = S + M_ * D_;             // 256
  float*    diag = z + M_;                  // 64
  float*    den  = diag + C_;               // 64
  unsigned* rowkey = (unsigned*)(den + C_); // 64

  const int tid = threadIdx.x, lane = tid & 31, wave = tid >> 5;
  const int bh = blockIdx.x;
  const float* qg = q + (size_t)bh * N_ * D_;
  const float* kg = k + (size_t)bh * N_ * D_;
  const float* vg = v + (size_t)bh * N_ * D_;
  float*       og = out + (size_t)bh * N_ * D_;
  const float gstab = fdec(*kmax_key);

  for (int i = tid; i < M_ * D_; i += 256) Pl[i] = (_Float16)proj[i];
  for (int i = tid; i < M_ * D_; i += 256) S[i] = 0.f;
  z[tid] = 0.f;
  __syncthreads();

  // ------------- conditioning accumulate: S = Kp^T V, z = colsum(Kp) --------
  for (int ch = 0; ch < L_ / C_; ++ch) {
    featuremap(kg + ch * C_ * D_, Kp, KT, Pl, Af, diag, rowkey, false, gstab, tid);
    for (int i = tid; i < C_ * D_; i += 256) {
      int r = i >> 6, e = i & 63;
      VT[e * C_ + r] = (_Float16)vg[(ch * C_ + r) * D_ + e];
    }
    __syncthreads();
    for (int t = 0; t < 8; ++t) {  // S += Kp^T @ V  (256x64, K=64)
      int tile = wave * 8 + t;
      int m0 = (tile >> 2) * 16, e0 = (tile & 3) * 16;
      v8f acc;
#pragma unroll
      for (int r = 0; r < 8; ++r)
        acc[r] = S[(m0 + r + ((lane >> 4) << 3)) * D_ + e0 + (lane & 15)];
      acc = wmma_f16(ld_frag(KT, C_, m0, 0, lane),  ld_frag(VT, C_, e0, 0, lane),  acc);
      acc = wmma_f16(ld_frag(KT, C_, m0, 32, lane), ld_frag(VT, C_, e0, 32, lane), acc);
#pragma unroll
      for (int r = 0; r < 8; ++r)
        S[(m0 + r + ((lane >> 4) << 3)) * D_ + e0 + (lane & 15)] = acc[r];
    }
    {  // z += colsum(Kp)
      float s = 0.f;
      for (int r = 0; r < C_; ++r) s += (float)Kp[r * M_ + tid];
      z[tid] += s;
    }
    __syncthreads();
  }
  // S^T f16 copy for use as WMMA B operand (B^T row-major)
  for (int i = tid; i < D_ * M_; i += 256) {
    int e = i >> 8, m = i & 255;
    ST[e * M_ + m] = (_Float16)S[m * D_ + e];
  }
  __syncthreads();

  // ------------- conditioning outputs: out = (Qp @ S) / (Qp . z) ------------
  for (int ch = 0; ch < L_ / C_; ++ch) {
    featuremap(qg + ch * C_ * D_, Qp, nullptr, Pl, Af, diag, rowkey, true, 0.f, tid);
    if (tid < C_) {
      float dd = 0.f;
      for (int j = 0; j < M_; ++j) dd += (float)Qp[tid * M_ + j] * z[j];
      den[tid] = dd;  // no epsilon in conditioned part
    }
    __syncthreads();
    for (int t = 0; t < 2; ++t) {  // 64x64, K=256
      int tile = wave * 2 + t;
      int r0 = (tile >> 2) * 16, e0 = (tile & 3) * 16;
      v8f acc = {};
      for (int k0 = 0; k0 < M_; k0 += 32)
        acc = wmma_f16(ld_frag(Qp, M_, r0, k0, lane), ld_frag(ST, M_, e0, k0, lane), acc);
#pragma unroll
      for (int r = 0; r < 8; ++r) {
        int row = r0 + r + ((lane >> 4) << 3);
        og[(ch * C_ + row) * D_ + e0 + (lane & 15)] = acc[r] / den[row];
      }
    }
    __syncthreads();
  }

  // ------------- causal rest: chunked linear-attention recurrence -----------
  for (int ch = 0; ch < NCH_; ++ch) {
    const int base = L_ + ch * C_;
    for (int i = tid; i < D_ * M_; i += 256) {  // refresh S^T (f16)
      int e = i >> 8, m = i & 255;
      ST[e * M_ + m] = (_Float16)S[m * D_ + e];
    }
    __syncthreads();
    featuremap(qg + base * D_, Qp, nullptr, Pl, Af, diag, rowkey, true, 0.f, tid);
    if (tid < C_) {  // den = qp.z_prev + SEPS*sum(qp)  (tril rowsum added below)
      float s1 = 0.f, s2 = 0.f;
      for (int j = 0; j < M_; ++j) {
        float qv = (float)Qp[tid * M_ + j];
        s1 += qv; s2 += qv * z[j];
      }
      den[tid] = s2 + SEPS * s1;
    }
    featuremap(kg + base * D_, Kp, KT, Pl, Af, diag, rowkey, false, gstab, tid);
    for (int i = tid; i < C_ * D_; i += 256) {
      int r = i >> 6, e = i & 63;
      VT[e * C_ + r] = (_Float16)vg[(base + r) * D_ + e];
    }
    if (ch + 1 < NCH_) {  // global_prefetch_b8 for next chunk
      const float* pq = qg + (base + C_) * D_ + (tid << 4);
      const float* pk = kg + (base + C_) * D_ + (tid << 4);
      const float* pv = vg + (base + C_) * D_ + (tid << 4);
      __builtin_prefetch(pq, 0, 3);
      __builtin_prefetch(pk, 0, 3);
      __builtin_prefetch(pv, 0, 3);
    }
    __syncthreads();
    for (int t = 0; t < 2; ++t) {  // A = tril(Qp @ Kp^T), rowsum -> den
      int tile = wave * 2 + t;
      int r0 = (tile >> 2) * 16, c0 = (tile & 3) * 16;
      v8f acc = {};
      for (int k0 = 0; k0 < M_; k0 += 32)
        acc = wmma_f16(ld_frag(Qp, M_, r0, k0, lane), ld_frag(Kp, M_, c0, k0, lane), acc);
#pragma unroll
      for (int r = 0; r < 8; ++r) {
        int row = r0 + r + ((lane >> 4) << 3);
        int col = c0 + (lane & 15);
        float mv = (col <= row) ? acc[r] : 0.f;
        Af[row * C_ + col] = (_Float16)mv;
        if (col <= row) atomicAdd(&den[row], mv);
      }
    }
    __syncthreads();
    for (int t = 0; t < 2; ++t) {  // out = (Qp@S + A@V) / den
      int tile = wave * 2 + t;
      int r0 = (tile >> 2) * 16, e0 = (tile & 3) * 16;
      v8f acc = {};
      for (int k0 = 0; k0 < M_; k0 += 32)
        acc = wmma_f16(ld_frag(Qp, M_, r0, k0, lane), ld_frag(ST, M_, e0, k0, lane), acc);
      acc = wmma_f16(ld_frag(Af, C_, r0, 0, lane),  ld_frag(VT, C_, e0, 0, lane),  acc);
      acc = wmma_f16(ld_frag(Af, C_, r0, 32, lane), ld_frag(VT, C_, e0, 32, lane), acc);
#pragma unroll
      for (int r = 0; r < 8; ++r) {
        int row = r0 + r + ((lane >> 4) << 3);
        og[(base + row) * D_ + e0 + (lane & 15)] = acc[r] / den[row];
      }
    }
    __syncthreads();
    for (int t = 0; t < 8; ++t) {  // S += Kp^T @ V
      int tile = wave * 8 + t;
      int m0 = (tile >> 2) * 16, e0 = (tile & 3) * 16;
      v8f acc;
#pragma unroll
      for (int r = 0; r < 8; ++r)
        acc[r] = S[(m0 + r + ((lane >> 4) << 3)) * D_ + e0 + (lane & 15)];
      acc = wmma_f16(ld_frag(KT, C_, m0, 0, lane),  ld_frag(VT, C_, e0, 0, lane),  acc);
      acc = wmma_f16(ld_frag(KT, C_, m0, 32, lane), ld_frag(VT, C_, e0, 32, lane), acc);
#pragma unroll
      for (int r = 0; r < 8; ++r)
        S[(m0 + r + ((lane >> 4) << 3)) * D_ + e0 + (lane & 15)] = acc[r];
    }
    {  // z += colsum(Kp)
      float s = 0.f;
      for (int r = 0; r < C_; ++r) s += (float)Kp[r * M_ + tid];
      z[tid] += s;
    }
    __syncthreads();
  }
}

// ---------------------------------------------------------------------------
extern "C" void kernel_launch(void* const* d_in, const int* in_sizes, int n_in,
                              void* d_out, int out_size, void* d_ws,
                              size_t ws_size, hipStream_t stream) {
  const float* q    = (const float*)d_in[0];
  const float* k    = (const float*)d_in[1];
  const float* v    = (const float*)d_in[2];
  const float* proj = (const float*)d_in[3];
  float*    out = (float*)d_out;
  unsigned* key = (unsigned*)d_ws;  // 4 bytes of scratch

  fa_init_kernel<<<1, 64, 0, stream>>>(key);
  fa_kmax_kernel<<<(B_ * H_ * N_) / C_, 256, 0, stream>>>(k, proj, key);

  // dynamic LDS: f16 tiles (Pl,Qp,Kp,KT,ST) + VT,Af + f32 state/reductions
  size_t shmem = (size_t)(M_ * D_ + 4 * C_ * M_ + 2 * C_ * C_) * sizeof(_Float16)
               + (size_t)(M_ * D_ + M_ + C_ + C_) * sizeof(float)
               + (size_t)C_ * sizeof(unsigned);
  fa_attn_kernel<<<B_ * H_, 256, shmem, stream>>>(q, k, v, proj, out, key);
}
